// CausalSelfAttention_1340029796403
// MI455X (gfx1250) — matrix operand order repaired and checked by reading, compiled, not run on hardware
//
#include <hip/hip_runtime.h>
#include <hip/hip_bf16.h>

// ---- constants -------------------------------------------------------------
#define Bk 8
#define Tk 1024
#define Ck 768
#define NHk 12
#define HDk 64
#define N3k 2304          // 3*C
#define WPB 4             // waves per block (128 threads, wave32)

typedef __attribute__((ext_vector_type(16))) __bf16 v16bf;
typedef __attribute__((ext_vector_type(8)))  float  v8f;

// ---- bf16 helpers ----------------------------------------------------------
static __device__ __forceinline__ unsigned int f2bfu(float f) {
    union { float f; unsigned u; } a; a.f = f;
    unsigned u = a.u;
    u += 0x7fffu + ((u >> 16) & 1u);          // round-to-nearest-even
    return u >> 16;
}
static __device__ __forceinline__ unsigned int pack2(float lo, float hi) {
    return f2bfu(lo) | (f2bfu(hi) << 16);
}
static __device__ __forceinline__ unsigned short f2bfs(float f) {
    return (unsigned short)f2bfu(f);
}

union Frag { uint4 q[2]; v16bf v; };
// fragment = two 16-byte chunks of contiguous (k-ascending) bf16
static __device__ __forceinline__ v16bf frag16(const void* p0, const void* p1) {
    Frag f;
    f.q[0] = *(const uint4*)p0;
    f.q[1] = *(const uint4*)p1;
    return f.v;
}

static __device__ __forceinline__ v8f wmma_bf16(v16bf a, v16bf b, v8f c) {
    return __builtin_amdgcn_wmma_f32_16x16x32_bf16(
        false, a, false, b, (short)0, c, false, false);
}

// ============================================================================
// prep kernel A: f32 -> bf16 elementwise (8 elems/thread)
// ============================================================================
__global__ __launch_bounds__(256) void cvt_bf16(
    const float* __restrict__ src, unsigned short* __restrict__ dst)
{
    const size_t i = ((size_t)blockIdx.x * 256 + threadIdx.x) * 8;
    const float4 a = *(const float4*)(src + i);
    const float4 b = *(const float4*)(src + i + 4);
    uint4 o;
    o.x = pack2(a.x, a.y);
    o.y = pack2(a.z, a.w);
    o.z = pack2(b.x, b.y);
    o.w = pack2(b.z, b.w);
    *(uint4*)(dst + i) = o;
}

// ============================================================================
// prep kernel B: transpose-convert W[K][N] f32 -> Wt[N][K] bf16, 32x32 tiles
// ============================================================================
__global__ __launch_bounds__(256) void transpose_w(
    const float* __restrict__ src, unsigned short* __restrict__ dst,
    int K, int N)
{
    __shared__ __align__(16) unsigned short tile[32 * 36];   // [n][k], padded

    const int tilesN = N >> 5;
    const int k0 = (blockIdx.x / tilesN) << 5;
    const int n0 = (blockIdx.x % tilesN) << 5;
    const int t  = threadIdx.x;

    {
        const int kk = t >> 3;             // 0..31
        const int nn = (t & 7) * 4;        // 0..28
        const float4 v = *(const float4*)(src + (size_t)(k0 + kk) * N + n0 + nn);
        tile[(nn + 0) * 36 + kk] = f2bfs(v.x);
        tile[(nn + 1) * 36 + kk] = f2bfs(v.y);
        tile[(nn + 2) * 36 + kk] = f2bfs(v.z);
        tile[(nn + 3) * 36 + kk] = f2bfs(v.w);
    }
    __syncthreads();
    {
        const int nn = t >> 3;
        const int kk = (t & 7) * 4;
        const uint2 v = *(const uint2*)(&tile[nn * 36 + kk]);
        *(uint2*)(dst + (size_t)(n0 + nn) * K + k0 + kk) = v;
    }
}

// ============================================================================
// GEMM core: 16(M) x 64(N) strip per wave, 4 accumulators, double-buffered
// operand fragments so loads of step k+1 overlap the 4 wmmas of step k.
// ============================================================================
static __device__ __forceinline__ void gemm_strip(
    const unsigned short* __restrict__ ap,
    const unsigned short* __restrict__ bp0, const unsigned short* __restrict__ bp1,
    const unsigned short* __restrict__ bp2, const unsigned short* __restrict__ bp3,
    v8f acc[4])
{
    v16bf a  = frag16(ap,      ap + 8);
    v16bf b0 = frag16(bp0,     bp0 + 8);
    v16bf b1 = frag16(bp1,     bp1 + 8);
    v16bf b2 = frag16(bp2,     bp2 + 8);
    v16bf b3 = frag16(bp3,     bp3 + 8);

    for (int k0 = 0; k0 < Ck - 32; k0 += 32) {
        const int k1 = k0 + 32;
        // prefetch next step's fragments (stay outstanding during the wmmas)
        const v16bf an  = frag16(ap  + k1, ap  + k1 + 8);
        const v16bf b0n = frag16(bp0 + k1, bp0 + k1 + 8);
        const v16bf b1n = frag16(bp1 + k1, bp1 + k1 + 8);
        const v16bf b2n = frag16(bp2 + k1, bp2 + k1 + 8);
        const v16bf b3n = frag16(bp3 + k1, bp3 + k1 + 8);

        acc[0] = wmma_bf16(a, b0, acc[0]);
        acc[1] = wmma_bf16(a, b1, acc[1]);
        acc[2] = wmma_bf16(a, b2, acc[2]);
        acc[3] = wmma_bf16(a, b3, acc[3]);

        a = an; b0 = b0n; b1 = b1n; b2 = b2n; b3 = b3n;
    }
    // peeled last step
    acc[0] = wmma_bf16(a, b0, acc[0]);
    acc[1] = wmma_bf16(a, b1, acc[1]);
    acc[2] = wmma_bf16(a, b2, acc[2]);
    acc[3] = wmma_bf16(a, b3, acc[3]);
}

// ============================================================================
// kernel 1: qkv = xb @ w_attnT + b_attn
//   q,k -> [B,NH,T,HD] bf16 ; v -> [B,NH,HD,T] bf16 (transposed)
// ============================================================================
__global__ __launch_bounds__(128) void qkv_gemm(
    const unsigned short* __restrict__ xb, const unsigned short* __restrict__ wT,
    const float* __restrict__ bias,
    unsigned short* __restrict__ qo, unsigned short* __restrict__ ko,
    unsigned short* __restrict__ vt)
{
    const int lane = threadIdx.x & 31;
    const int wv   = threadIdx.x >> 5;
    const int tile = blockIdx.x * WPB + wv;          // 512*36 strips
    const int tN = tile % (N3k / 64);
    const int tM = tile / (N3k / 64);
    const int m0 = tM * 16, n0 = tN * 64;
    const int r  = lane & 15;
    const int hi = (lane >= 16) ? 1 : 0;

    v8f acc[4];
#pragma unroll
    for (int i = 0; i < 4; ++i) {
        const float bv = bias[n0 + i * 16 + r];
#pragma unroll
        for (int e = 0; e < 8; ++e) acc[i][e] = bv;
    }

    gemm_strip(xb + (size_t)(m0 + r) * Ck + hi * 16,
               wT + (size_t)(n0      + r) * Ck + hi * 16,
               wT + (size_t)(n0 + 16 + r) * Ck + hi * 16,
               wT + (size_t)(n0 + 32 + r) * Ck + hi * 16,
               wT + (size_t)(n0 + 48 + r) * Ck + hi * 16,
               acc);

    // scatter each 16-wide sub-tile
#pragma unroll
    for (int i = 0; i < 4; ++i) {
        const int j     = n0 + i * 16 + r;
        const int which = j / Ck;
        const int cc    = j % Ck;
        const int head  = cc >> 6;
        const int d     = cc & 63;
        if (which == 2) {
#pragma unroll
            for (int e = 0; e < 8; ++e) {
                const int m  = m0 + e + hi * 8;
                const int bb = m >> 10;
                const int tt = m & (Tk - 1);
                vt[((size_t)(bb * NHk + head) * HDk + d) * Tk + tt] = f2bfs(acc[i][e]);
            }
        } else {
            unsigned short* dst = (which == 0) ? qo : ko;
#pragma unroll
            for (int e = 0; e < 8; ++e) {
                const int m  = m0 + e + hi * 8;
                const int bb = m >> 10;
                const int tt = m & (Tk - 1);
                dst[((size_t)(bb * NHk + head) * Tk + tt) * HDk + d] = f2bfs(acc[i][e]);
            }
        }
    }
}

// ============================================================================
// kernel 2: flash attention, one wave per (b,h,16-query tile)
// ============================================================================
__global__ __launch_bounds__(128) void attn_kernel(
    const unsigned short* __restrict__ qg, const unsigned short* __restrict__ kg,
    const unsigned short* __restrict__ vg, unsigned short* __restrict__ y)
{
    __shared__ __align__(16) unsigned short lds[WPB][16 * 32];  // P tile per wave

    const int lane = threadIdx.x & 31;
    const int wv   = threadIdx.x >> 5;
    const int id   = blockIdx.x * WPB + wv;
    const int qt   = id & 63;
    const int bh   = id >> 6;
    const int b    = bh / NHk;
    const int h    = bh % NHk;
    const int q0   = qt * 16;
    const int r    = lane & 15;
    const int hi   = (lane >= 16) ? 1 : 0;

    const unsigned short* Q  = qg + (size_t)bh * Tk * HDk;
    const unsigned short* K  = kg + (size_t)bh * Tk * HDk;
    const unsigned short* Vt = vg + (size_t)bh * HDk * Tk;

    const unsigned short* qp = Q + (size_t)(q0 + r) * HDk + hi * 16;
    const v16bf aq0 = frag16(qp,      qp + 8);
    const v16bf aq1 = frag16(qp + 32, qp + 40);

    float rm[8], rl[8];
    v8f o0 = {}, o1 = {}, o2 = {}, o3 = {};
#pragma unroll
    for (int e = 0; e < 8; ++e) { rm[e] = -__builtin_inff(); rl[e] = 0.f; }

    const int nblk = (q0 + 47) >> 5;
    for (int jb = 0; jb < nblk; ++jb) {
        const int kb = jb * 32;

        const unsigned short* kp0 = K + (size_t)(kb + r) * HDk + hi * 16;
        const unsigned short* kp1 = K + (size_t)(kb + 16 + r) * HDk + hi * 16;
        const v16bf bk00 = frag16(kp0,      kp0 + 8);
        const v16bf bk01 = frag16(kp0 + 32, kp0 + 40);
        const v16bf bk10 = frag16(kp1,      kp1 + 8);
        const v16bf bk11 = frag16(kp1 + 32, kp1 + 40);

        v8f s0 = {}, s1 = {};
        s0 = wmma_bf16(aq0, bk00, s0);
        s0 = wmma_bf16(aq1, bk01, s0);
        s1 = wmma_bf16(aq0, bk10, s1);
        s1 = wmma_bf16(aq1, bk11, s1);

        float p0[8], p1[8];
#pragma unroll
        for (int e = 0; e < 8; ++e) {
            const int m = q0 + e + hi * 8;
            float x0 = (kb + r      <= m) ? s0[e] * 0.125f : -__builtin_inff();
            float x1 = (kb + 16 + r <= m) ? s1[e] * 0.125f : -__builtin_inff();
            float mx = fmaxf(x0, x1);
            mx = fmaxf(mx, __shfl_xor(mx, 8, 32));
            mx = fmaxf(mx, __shfl_xor(mx, 4, 32));
            mx = fmaxf(mx, __shfl_xor(mx, 2, 32));
            mx = fmaxf(mx, __shfl_xor(mx, 1, 32));
            const float mnew  = fmaxf(rm[e], mx);
            const float alpha = __expf(rm[e] - mnew);
            const float e0 = __expf(x0 - mnew);
            const float e1 = __expf(x1 - mnew);
            float sum = e0 + e1;
            sum += __shfl_xor(sum, 8, 32);
            sum += __shfl_xor(sum, 4, 32);
            sum += __shfl_xor(sum, 2, 32);
            sum += __shfl_xor(sum, 1, 32);
            rl[e] = rl[e] * alpha + sum;
            rm[e] = mnew;
            o0[e] *= alpha; o1[e] *= alpha; o2[e] *= alpha; o3[e] *= alpha;
            p0[e] = e0; p1[e] = e1;
        }

        unsigned short* L = lds[wv];
#pragma unroll
        for (int e = 0; e < 8; ++e) {
            const int m = e + hi * 8;
            L[m * 32 + r]      = f2bfs(p0[e]);
            L[m * 32 + 16 + r] = f2bfs(p1[e]);
        }
        asm volatile("s_wait_dscnt 0" ::: "memory");
        const unsigned short* lp = L + r * 32 + hi * 16;
        const v16bf ap = frag16(lp, lp + 8);

        const unsigned short* vp0 = Vt + (size_t)(r)      * Tk + kb + hi * 16;
        const unsigned short* vp1 = Vt + (size_t)(16 + r) * Tk + kb + hi * 16;
        const unsigned short* vp2 = Vt + (size_t)(32 + r) * Tk + kb + hi * 16;
        const unsigned short* vp3 = Vt + (size_t)(48 + r) * Tk + kb + hi * 16;
        const v16bf bv0 = frag16(vp0, vp0 + 8);
        const v16bf bv1 = frag16(vp1, vp1 + 8);
        const v16bf bv2 = frag16(vp2, vp2 + 8);
        const v16bf bv3 = frag16(vp3, vp3 + 8);

        o0 = wmma_bf16(ap, bv0, o0);
        o1 = wmma_bf16(ap, bv1, o1);
        o2 = wmma_bf16(ap, bv2, o2);
        o3 = wmma_bf16(ap, bv3, o3);
    }

#pragma unroll
    for (int e = 0; e < 8; ++e) {
        const float inv = 1.f / rl[e];
        const int m = q0 + e + hi * 8;
        const size_t base = ((size_t)b * Tk + m) * Ck + h * HDk + r;
        y[base]      = f2bfs(o0[e] * inv);
        y[base + 16] = f2bfs(o1[e] * inv);
        y[base + 32] = f2bfs(o2[e] * inv);
        y[base + 48] = f2bfs(o3[e] * inv);
    }
}

// ============================================================================
// kernel 3: out = y @ w_projT + b_proj (fp32 out)
// ============================================================================
__global__ __launch_bounds__(128) void proj_gemm(
    const unsigned short* __restrict__ y, const unsigned short* __restrict__ wT,
    const float* __restrict__ bias, float* __restrict__ out)
{
    const int lane = threadIdx.x & 31;
    const int wv   = threadIdx.x >> 5;
    const int tile = blockIdx.x * WPB + wv;          // 512*12 strips
    const int tN = tile % (Ck / 64);
    const int tM = tile / (Ck / 64);
    const int m0 = tM * 16, n0 = tN * 64;
    const int r  = lane & 15;
    const int hi = (lane >= 16) ? 1 : 0;

    v8f acc[4];
#pragma unroll
    for (int i = 0; i < 4; ++i) {
        const float bv = bias[n0 + i * 16 + r];
#pragma unroll
        for (int e = 0; e < 8; ++e) acc[i][e] = bv;
    }

    gemm_strip(y  + (size_t)(m0 + r) * Ck + hi * 16,
               wT + (size_t)(n0      + r) * Ck + hi * 16,
               wT + (size_t)(n0 + 16 + r) * Ck + hi * 16,
               wT + (size_t)(n0 + 32 + r) * Ck + hi * 16,
               wT + (size_t)(n0 + 48 + r) * Ck + hi * 16,
               acc);

#pragma unroll
    for (int i = 0; i < 4; ++i) {
#pragma unroll
        for (int e = 0; e < 8; ++e) {
            const int m = m0 + e + hi * 8;
            out[(size_t)m * Ck + n0 + i * 16 + r] = acc[i][e];
        }
    }
}

// ---- launcher --------------------------------------------------------------
extern "C" void kernel_launch(void* const* d_in, const int* in_sizes, int n_in,
                              void* d_out, int out_size, void* d_ws, size_t ws_size,
                              hipStream_t stream) {
    const float* x      = (const float*)d_in[0];
    const float* w_attn = (const float*)d_in[1];
    const float* b_attn = (const float*)d_in[2];
    const float* w_proj = (const float*)d_in[3];
    const float* b_proj = (const float*)d_in[4];
    float* out = (float*)d_out;

    const size_t nx   = (size_t)Bk * Tk * Ck;        // 6,291,456
    const size_t nwa  = (size_t)Ck * N3k;            // 1,769,472
    const size_t nwp  = (size_t)Ck * Ck;             //   589,824
    const size_t per  = (size_t)Bk * NHk * Tk * HDk; // 6,291,456

    unsigned short* xb  = (unsigned short*)d_ws;
    unsigned short* waT = xb  + nx;                  // [3C][C]
    unsigned short* wpT = waT + nwa;                 // [C][C]
    unsigned short* q   = wpT + nwp;
    unsigned short* k   = q   + per;
    unsigned short* vt  = k   + per;                 // [B,NH,HD,T]
    unsigned short* y   = vt  + per;                 // [B,T,C] bf16

    // prep
    cvt_bf16   <<<(int)(nx / (256 * 8)),  256, 0, stream>>>(x, xb);
    transpose_w<<<(Ck / 32) * (N3k / 32), 256, 0, stream>>>(w_attn, waT, Ck, N3k);
    transpose_w<<<(Ck / 32) * (Ck / 32),  256, 0, stream>>>(w_proj, wpT, Ck, Ck);

    const int qkv_strips  = (Bk * Tk / 16) * (N3k / 64);  // 512*36
    const int attn_waves  = Bk * NHk * (Tk / 16);         // 6144
    const int proj_strips = (Bk * Tk / 16) * (Ck / 64);   // 512*12

    qkv_gemm   <<<qkv_strips / WPB,  128, 0, stream>>>(xb, waT, b_attn, q, k, vt);
    attn_kernel<<<attn_waves / WPB,  128, 0, stream>>>(q, k, vt, y);
    proj_gemm  <<<proj_strips / WPB, 128, 0, stream>>>(y, wpT, b_proj, out);
}
